// GraphSAGE_22351009808410
// MI455X (gfx1250) — matrix operand order repaired and checked by reading, not compiled, not run on hardware
//
#include <hip/hip_runtime.h>

typedef __attribute__((ext_vector_type(16))) _Float16 v16h;
typedef __attribute__((ext_vector_type(8)))  float    v8f;

#define N_NODES 50000
#define N_EDGES 800000
#define IN_CH   64
#define HID_CH  128
#define M_TILES (N_NODES / 16)   // 3125
#define N_TILES (HID_CH / 16)    // 8

// Workspace layout (d_ws is 256B aligned):
//   float    sums[N_NODES*IN_CH]   12.8 MB
//   float    counts[N_NODES]        0.2 MB
//   _Float16 Wh[HID_CH*IN_CH]      16 KB   (offset 13,000,000 B -> 16B aligned)

// Zero the accumulators AND pre-convert W (f32 -> f16) for async LDS staging.
__global__ void sage_zero_prep_kernel(float* __restrict__ ws, int total,
                                      const float* __restrict__ W,
                                      _Float16* __restrict__ Wh) {
    int i = blockIdx.x * blockDim.x + threadIdx.x;
    if (i < total) ws[i] = 0.0f;
    if (i < HID_CH * IN_CH) Wh[i] = (_Float16)W[i];
}

__global__ void sage_scatter_kernel(const float* __restrict__ x,
                                    const int* __restrict__ edges,  // [2, E] flat
                                    float* __restrict__ sums,
                                    float* __restrict__ counts) {
    int tid = blockIdx.x * blockDim.x + threadIdx.x;
    if (tid >= N_EDGES * IN_CH) return;
    int e = tid >> 6;          // edge id (32 lanes of a wave share one edge)
    int c = tid & (IN_CH - 1); // channel -> coalesced 128B feature reads
    int src = edges[e];
    int dst = edges[N_EDGES + e];
    float v = x[src * IN_CH + c];
    atomicAdd(&sums[dst * IN_CH + c], v);
    if (c == 0) atomicAdd(&counts[dst], 1.0f);
}

// Fused: mean (= sums * 1/max(cnt,1)) -> f16 A fragments in registers,
// then out[n,o] = sum_k mean[n,k] * W[o,k] via 2x v_wmma_f32_16x16x32_f16.
__global__ void __launch_bounds__(256)
sage_wmma_gemm_kernel(const float* __restrict__ sums,     // [N_NODES, IN_CH] f32
                      const float* __restrict__ counts,   // [N_NODES] f32
                      const _Float16* __restrict__ Wh,    // [HID_CH, IN_CH] f16
                      float* __restrict__ out) {          // [N_NODES, HID_CH] f32
    __shared__ __align__(16) _Float16 Wl[HID_CH * IN_CH]; // 16 KB

    int t = threadIdx.x;

    // ---- Async-stage W into LDS: 16 KB = 1024 x b128, 4 per thread ----
    {
        unsigned lds_base = (unsigned)(uintptr_t)(&Wl[0]); // low 32b of generic = LDS offset
        const char* gbase = (const char*)Wh;
        #pragma unroll
        for (int it = 0; it < 4; ++it) {
            int off = (t + it * 256) * 16; // bytes
            unsigned lds_off = lds_base + (unsigned)off;
            const void* gaddr = gbase + off;
            asm volatile("global_load_async_to_lds_b128 %0, %1, off"
                         :: "v"(lds_off), "v"(gaddr)
                         : "memory");
        }
        asm volatile("s_wait_asynccnt 0x0" ::: "memory");
    }
    __syncthreads();

    int wave = t >> 5;
    int lane = t & 31;
    int mTile = blockIdx.x * 8 + wave;
    if (mTile >= M_TILES) return;  // wave-uniform: EXEC stays all-1s for WMMA waves

    int row16 = lane & 15;   // M index within tile (A), N index (B/D)
    int hi    = lane >> 4;   // lane-half selector

    // ---- Build A fragments directly from f32 sums (fused mean + cvt) ----
    // A 16x32 f16 layout: lanes 0-15 hold K {0..7,16..23}, lanes 16-31 {8..15,24..31}.
    int m = mTile * 16 + row16;
    const float* srow = sums + (size_t)m * IN_CH;
    float rcp = 1.0f / fmaxf(counts[m], 1.0f);

    float4 f0 = *(const float4*)(srow + hi * 8);
    float4 f1 = *(const float4*)(srow + hi * 8 + 4);
    float4 f2 = *(const float4*)(srow + 16 + hi * 8);
    float4 f3 = *(const float4*)(srow + 16 + hi * 8 + 4);
    float4 f4 = *(const float4*)(srow + 32 + hi * 8);
    float4 f5 = *(const float4*)(srow + 32 + hi * 8 + 4);
    float4 f6 = *(const float4*)(srow + 48 + hi * 8);
    float4 f7 = *(const float4*)(srow + 48 + hi * 8 + 4);

    union AU { v16h v; _Float16 h[16]; uint4 q[2]; };
    AU a0, a1;
    a0.h[0]  = (_Float16)(f0.x * rcp); a0.h[1]  = (_Float16)(f0.y * rcp);
    a0.h[2]  = (_Float16)(f0.z * rcp); a0.h[3]  = (_Float16)(f0.w * rcp);
    a0.h[4]  = (_Float16)(f1.x * rcp); a0.h[5]  = (_Float16)(f1.y * rcp);
    a0.h[6]  = (_Float16)(f1.z * rcp); a0.h[7]  = (_Float16)(f1.w * rcp);
    a0.h[8]  = (_Float16)(f2.x * rcp); a0.h[9]  = (_Float16)(f2.y * rcp);
    a0.h[10] = (_Float16)(f2.z * rcp); a0.h[11] = (_Float16)(f2.w * rcp);
    a0.h[12] = (_Float16)(f3.x * rcp); a0.h[13] = (_Float16)(f3.y * rcp);
    a0.h[14] = (_Float16)(f3.z * rcp); a0.h[15] = (_Float16)(f3.w * rcp);
    a1.h[0]  = (_Float16)(f4.x * rcp); a1.h[1]  = (_Float16)(f4.y * rcp);
    a1.h[2]  = (_Float16)(f4.z * rcp); a1.h[3]  = (_Float16)(f4.w * rcp);
    a1.h[4]  = (_Float16)(f5.x * rcp); a1.h[5]  = (_Float16)(f5.y * rcp);
    a1.h[6]  = (_Float16)(f5.z * rcp); a1.h[7]  = (_Float16)(f5.w * rcp);
    a1.h[8]  = (_Float16)(f6.x * rcp); a1.h[9]  = (_Float16)(f6.y * rcp);
    a1.h[10] = (_Float16)(f6.z * rcp); a1.h[11] = (_Float16)(f6.w * rcp);
    a1.h[12] = (_Float16)(f7.x * rcp); a1.h[13] = (_Float16)(f7.y * rcp);
    a1.h[14] = (_Float16)(f7.z * rcp); a1.h[15] = (_Float16)(f7.w * rcp);

    // ---- Sweep all 8 N-tiles, reusing A fragments ----
    for (int nt = 0; nt < N_TILES; ++nt) {
        int o = nt * 16 + row16;  // output channel = B column
        const _Float16* wrow = Wl + o * IN_CH;
        // B 32x16 f16 layout: lane holds column N=lane%16; lanes 0-15 hold
        // K {0..7,16..23}, lanes 16-31 hold K {8..15,24..31} (matches A).
        AU b0, b1;
        b0.q[0] = *(const uint4*)(wrow + hi * 8);
        b0.q[1] = *(const uint4*)(wrow + 16 + hi * 8);
        b1.q[0] = *(const uint4*)(wrow + 32 + hi * 8);
        b1.q[1] = *(const uint4*)(wrow + 48 + hi * 8);

        v8f c = {};
        c = __builtin_amdgcn_wmma_f32_16x16x32_f16(false, a0.v, false, b0.v,
                                                   (short)0, c, false, false);
        c = __builtin_amdgcn_wmma_f32_16x16x32_f16(false, a1.v, false, b1.v,
                                                   (short)0, c, false, false);

        // D layout: VGPR r -> row M = r + 8*hi, col N = lane%16.
        int obase = nt * 16 + row16;
        #pragma unroll
        for (int r = 0; r < 8; ++r) {
            out[(size_t)(mTile * 16 + hi * 8 + r) * HID_CH + obase] = c[r];
        }
    }
}

extern "C" void kernel_launch(void* const* d_in, const int* in_sizes, int n_in,
                              void* d_out, int out_size, void* d_ws, size_t ws_size,
                              hipStream_t stream) {
    const float* x     = (const float*)d_in[0];
    const int*   edges = (const int*)d_in[1];   // [2, E] int32
    const float* W     = (const float*)d_in[2]; // [HID_CH, IN_CH]

    float* sums   = (float*)d_ws;
    float* counts = sums + (size_t)N_NODES * IN_CH;
    _Float16* Wh  = (_Float16*)(counts + N_NODES);

    float* out = (float*)d_out;

    // 1) zero accumulators + pre-convert W to f16
    {
        int total = N_NODES * IN_CH + N_NODES;
        int blocks = (total + 255) / 256;
        sage_zero_prep_kernel<<<blocks, 256, 0, stream>>>((float*)d_ws, total, W, Wh);
    }
    // 2) edge scatter (atomic f32 accumulate into L2-resident sums)
    {
        int total = N_EDGES * IN_CH;          // 51.2M threads
        int blocks = (total + 255) / 256;     // 200000 blocks
        sage_scatter_kernel<<<blocks, 256, 0, stream>>>(x, edges, sums, counts);
    }
    // 3) fused mean + WMMA GEMM: [50000,64] x [64,128] -> [50000,128]
    {
        int blocks = (M_TILES + 7) / 8;       // 391 blocks x 8 waves
        sage_wmma_gemm_kernel<<<blocks, 256, 0, stream>>>(sums, counts, Wh, out);
    }
}
